// Model_36696200577167
// MI455X (gfx1250) — compile-verified, matrix operand order checked by reading, not run on hardware
//
#include <hip/hip_runtime.h>
#include <hip/hip_bf16.h>
#include <stdint.h>

// Problem dimensions (fixed by the reference).
#define B_   32
#define W_   168
#define M_   2048
#define HID_ 128
#define NL_  6
#define BW_  (B_ * W_)   // 5376

typedef __attribute__((ext_vector_type(16))) __bf16 bf16x16;
typedef __attribute__((ext_vector_type(8)))  float  f32x8;
typedef int v4i_vs __attribute__((vector_size(16)));   // matches builtin's V4i

#define AS1 __attribute__((address_space(1)))
#define AS3 __attribute__((address_space(3)))

#if __has_builtin(__builtin_amdgcn_global_load_async_to_lds_b128) && \
    __has_builtin(__builtin_amdgcn_s_wait_asynccnt)
#define ASYNC_OK 1
#else
#define ASYNC_OK 0
#endif

// ---------- bf16 split helpers (bit-level round-to-nearest-even) ----------
__device__ __forceinline__ unsigned short bf16_rne_bits(float f) {
    unsigned int u = __float_as_uint(f);
    unsigned int r = u + 0x7fffu + ((u >> 16) & 1u);
    return (unsigned short)(r >> 16);
}
__device__ __forceinline__ float bf16_bits_to_f(unsigned short h) {
    return __uint_as_float(((unsigned int)h) << 16);
}
__device__ __forceinline__ void split2(float f, unsigned short& hi, unsigned short& lo) {
    hi = bf16_rne_bits(f);
    lo = bf16_rne_bits(f - bf16_bits_to_f(hi));
}

// async 16B global -> LDS (per-lane addresses); guarded fallback never calls this
__device__ __forceinline__ void async_ld16(const unsigned short* g, unsigned short* l) {
#if ASYNC_OK
    __builtin_amdgcn_global_load_async_to_lds_b128(
        (AS1 v4i_vs*)(uintptr_t)(g),            // flat VA == global VA
        (AS3 v4i_vs*)(uintptr_t)(l),            // low 32 bits of generic LDS ptr = LDS offset
        0, 0);
#else
    (void)g; (void)l;
#endif
}

// ---------- split x[B,W,M] into xflat (=[B*W, M]) and xcat (=[M, B*W]) ----------
__global__ void k_split_x(const float* __restrict__ x,
                          unsigned short* __restrict__ xf_hi, unsigned short* __restrict__ xf_lo,
                          unsigned short* __restrict__ xc_hi, unsigned short* __restrict__ xc_lo) {
    int i = blockIdx.x * 256 + threadIdx.x;
    if (i >= BW_ * M_) return;
    int bw = i / M_;
    int m  = i - bw * M_;
    unsigned short h, l;
    split2(x[i], h, l);
    xf_hi[i] = h;  xf_lo[i] = l;
    long ti = (long)m * BW_ + bw;
    xc_hi[ti] = h;  xc_lo[ti] = l;
}

// ---------- generic f32 -> bf16 hi/lo split, optional transposed copy (setup only) ----------
__global__ void k_split(const float* __restrict__ src,
                        unsigned short* __restrict__ hi, unsigned short* __restrict__ lo,
                        unsigned short* __restrict__ hiT, unsigned short* __restrict__ loT,
                        int R, int C) {
    int i = blockIdx.x * 256 + threadIdx.x;
    if (i >= R * C) return;
    int r = i / C, c = i - r * C;
    unsigned short h, l;
    split2(src[i], h, l);
    hi[i] = h;  lo[i] = l;
    if (hiT) {
        long ti = (long)c * R + r;
        hiT[ti] = h;  loT[ti] = l;
    }
}

// ---------- bf16x3 split GEMM ----------
// acc[m,n] = sum_k (Ahi+Alo)[m,k] * (Bthi+Btlo)[n,k], fp32 accumulate.
// Block 256 threads / 8 waves (4 M-strips x 2 N-strips), BM=256 x BN=128, BK=32,
// each wave computes 64x64 (4x4 WMMA tiles). Double-buffered LDS, 1 barrier/k-step.
// Epilogue modes:
//   0: emit bf16 hi/lo of result (Chi/Clo, ldcb), LDS-staged coalesced stores
//   1: mode 0 + transposed bf16 hi/lo (Cthi/Ctlo, ldct), LDS-transposed coalesced stores
//   2: fused MLP: s[z*s_zoff+row] (+)= sum_n W2[hop][n]*relu(acc+b1[hop][n]) + b2[hop]
#define BM 256
#define BN 128
#define BK 32
#define LDST 40    // LDS staging row stride (halves): 16 rows cover all 64 banks
#define ESTR (BN + 8)   // epilogue plain-stage stride  [BM][136]
#define TSTR (BM + 8)   // epilogue transpose-stage stride [BN][264]

union ABReg { bf16x16 v; uint4 q[2]; };
union CReg  { f32x8 v; float f[8]; };

__device__ __forceinline__ uint4 load8_guard(const unsigned short* __restrict__ p,
                                             long off, int kbase, int Kdim) {
    if (kbase + 8 <= Kdim) return *(const uint4*)(p + off);
    union { uint4 q; unsigned short h[8]; } u;
    #pragma unroll
    for (int e = 0; e < 8; ++e)
        u.h[e] = (kbase + e < Kdim) ? p[off + e] : (unsigned short)0;
    return u.q;
}

template<bool KTAIL>
__global__ __launch_bounds__(256)
void k_gemm(const unsigned short* __restrict__ Ahi, const unsigned short* __restrict__ Alo,
            long lda, long a_zoff,
            const unsigned short* __restrict__ Bthi, const unsigned short* __restrict__ Btlo,
            long ldbt, int Kdim, int mode,
            unsigned short* __restrict__ Chi, unsigned short* __restrict__ Clo, long ldcb,
            unsigned short* __restrict__ Cthi, unsigned short* __restrict__ Ctlo, long ldct,
            const float* __restrict__ b1, const float* __restrict__ W2,
            const float* __restrict__ b2, float* __restrict__ sOut, long s_zoff,
            int hop, int accumulate) {
    __shared__ __align__(16) unsigned short sA_hi[2][BM * LDST];
    __shared__ __align__(16) unsigned short sA_lo[2][BM * LDST];
    __shared__ __align__(16) unsigned short sB_hi[2][BN * LDST];
    __shared__ __align__(16) unsigned short sB_lo[2][BN * LDST];
    __shared__ float sred[2][BM];

    const int tid  = threadIdx.x;
    const int lane = tid & 31;
    const int wave = tid >> 5;
    const int wm   = wave >> 1;          // 0..3 : 64-row strip
    const int wn   = wave & 1;           // 0..1 : 64-col strip
    const int row0 = blockIdx.y * BM;
    const int col0 = blockIdx.x * BN;
    const long z   = blockIdx.z;

    const unsigned short* __restrict__ Ah = Ahi + z * a_zoff;
    const unsigned short* __restrict__ Al = Alo + z * a_zoff;
    const unsigned short* __restrict__ Bh = Bthi;
    const unsigned short* __restrict__ Bl = Btlo;

    f32x8 acc[4][4];
    {
        f32x8 zv = {};
        #pragma unroll
        for (int i = 0; i < 4; ++i)
            #pragma unroll
            for (int j = 0; j < 4; ++j) acc[i][j] = zv;
    }

    // Staging map: thread t owns A row (row0+t), 32 cols; Bt row (col0+t/2), 16 cols.
    const long a_gbase = (long)(row0 + tid) * lda;
    const int  b_cb    = (tid & 1) * 16;
    const long b_gbase = (long)(col0 + (tid >> 1)) * ldbt + b_cb;

    const int khalf = lane >> 4;  // 0/1
    const int lm    = lane & 15;

    const int nk = (Kdim + BK - 1) / BK;

    auto compute_tile = [&](int buf) {
        ABReg a_h[4], a_l[4];
        #pragma unroll
        for (int mt = 0; mt < 4; ++mt) {
            const int m = wm * 64 + mt * 16 + lm;
            const unsigned short* ph = sA_hi[buf] + m * LDST;
            const unsigned short* pl = sA_lo[buf] + m * LDST;
            a_h[mt].q[0] = *(const uint4*)(ph + khalf * 8);
            a_h[mt].q[1] = *(const uint4*)(ph + 16 + khalf * 8);
            a_l[mt].q[0] = *(const uint4*)(pl + khalf * 8);
            a_l[mt].q[1] = *(const uint4*)(pl + 16 + khalf * 8);
        }
        #pragma unroll
        for (int nt = 0; nt < 4; ++nt) {
            const int n = wn * 64 + nt * 16 + lm;
            const unsigned short* ph = sB_hi[buf] + n * LDST;
            const unsigned short* pl = sB_lo[buf] + n * LDST;
            ABReg b_h, b_l;
            b_h.q[0] = *(const uint4*)(ph + khalf * 8);
            b_h.q[1] = *(const uint4*)(ph + 16 + khalf * 8);
            b_l.q[0] = *(const uint4*)(pl + khalf * 8);
            b_l.q[1] = *(const uint4*)(pl + 16 + khalf * 8);
            #pragma unroll
            for (int mt = 0; mt < 4; ++mt) {
                acc[mt][nt] = __builtin_amdgcn_wmma_f32_16x16x32_bf16(
                    false, a_h[mt].v, false, b_h.v, (short)0, acc[mt][nt], false, false);
                acc[mt][nt] = __builtin_amdgcn_wmma_f32_16x16x32_bf16(
                    false, a_h[mt].v, false, b_l.v, (short)0, acc[mt][nt], false, false);
                acc[mt][nt] = __builtin_amdgcn_wmma_f32_16x16x32_bf16(
                    false, a_l[mt].v, false, b_h.v, (short)0, acc[mt][nt], false, false);
            }
        }
    };

#if ASYNC_OK
    if constexpr (!KTAIL) {
        // ---- async global->LDS staging pipeline ----
        auto issue_tile = [&](int j, int buf) {
            const int k0 = j * BK;
            unsigned short* da_h = sA_hi[buf] + tid * LDST;
            unsigned short* da_l = sA_lo[buf] + tid * LDST;
            #pragma unroll
            for (int c = 0; c < 4; ++c) {
                async_ld16(Ah + a_gbase + k0 + c * 8, da_h + c * 8);
                async_ld16(Al + a_gbase + k0 + c * 8, da_l + c * 8);
            }
            unsigned short* db_h = sB_hi[buf] + (tid >> 1) * LDST + b_cb;
            unsigned short* db_l = sB_lo[buf] + (tid >> 1) * LDST + b_cb;
            #pragma unroll
            for (int c = 0; c < 2; ++c) {
                async_ld16(Bh + b_gbase + k0 + c * 8, db_h + c * 8);
                async_ld16(Bl + b_gbase + k0 + c * 8, db_l + c * 8);
            }
        };
        issue_tile(0, 0);
        for (int j = 0; j < nk; ++j) {
            __builtin_amdgcn_s_wait_asynccnt(0);   // own tile-j loads landed
            __syncthreads();                       // everyone's landed; prev reads done
            if (j + 1 < nk) issue_tile(j + 1, (j + 1) & 1);
            compute_tile(j & 1);
        }
    } else
#endif
    {
        // ---- register-staged pipeline (fallback / K-tail) ----
        uint4 ra_h[4], ra_l[4], rb_h[2], rb_l[2];
        auto fetch_tile = [&](int j) {
            const int k0 = j * BK;
            #pragma unroll
            for (int c = 0; c < 4; ++c) {
                const int kb = k0 + c * 8;
                if (KTAIL) {
                    ra_h[c] = load8_guard(Ah, a_gbase + kb, kb, Kdim);
                    ra_l[c] = load8_guard(Al, a_gbase + kb, kb, Kdim);
                } else {
                    ra_h[c] = *(const uint4*)(Ah + a_gbase + kb);
                    ra_l[c] = *(const uint4*)(Al + a_gbase + kb);
                }
            }
            #pragma unroll
            for (int c = 0; c < 2; ++c) {
                const int kb = k0 + b_cb + c * 8;
                if (KTAIL) {
                    rb_h[c] = load8_guard(Bh, b_gbase + k0 + c * 8, kb, Kdim);
                    rb_l[c] = load8_guard(Bl, b_gbase + k0 + c * 8, kb, Kdim);
                } else {
                    rb_h[c] = *(const uint4*)(Bh + b_gbase + k0 + c * 8);
                    rb_l[c] = *(const uint4*)(Bl + b_gbase + k0 + c * 8);
                }
            }
        };
        fetch_tile(0);
        for (int j = 0; j < nk; ++j) {
            const int buf = j & 1;
            {
                unsigned short* da_h = sA_hi[buf] + tid * LDST;
                unsigned short* da_l = sA_lo[buf] + tid * LDST;
                #pragma unroll
                for (int c = 0; c < 4; ++c) {
                    *(uint4*)(da_h + c * 8) = ra_h[c];
                    *(uint4*)(da_l + c * 8) = ra_l[c];
                }
                unsigned short* db_h = sB_hi[buf] + (tid >> 1) * LDST + b_cb;
                unsigned short* db_l = sB_lo[buf] + (tid >> 1) * LDST + b_cb;
                #pragma unroll
                for (int c = 0; c < 2; ++c) {
                    *(uint4*)(db_h + c * 8) = rb_h[c];
                    *(uint4*)(db_l + c * 8) = rb_l[c];
                }
            }
            if (j + 1 < nk) fetch_tile(j + 1);
            __syncthreads();
            compute_tile(buf);
        }
    }

    // ================= epilogue =================
    if (mode == 2) {
        // Fused MLP: partial over own n-columns, butterfly over the 16-lane n-group,
        // cross-wave combine in LDS, then s (+)= total + b2.
        float p[4][8];
        #pragma unroll
        for (int mt = 0; mt < 4; ++mt)
            #pragma unroll
            for (int r = 0; r < 8; ++r) p[mt][r] = 0.f;
        #pragma unroll
        for (int mt = 0; mt < 4; ++mt)
            #pragma unroll
            for (int nt = 0; nt < 4; ++nt) {
                CReg cr; cr.v = acc[mt][nt];
                const int n = wn * 64 + nt * 16 + lm;   // col0 == 0 in fused mode
                const float bb = b1[hop * HID_ + n];
                const float w2 = W2[hop * HID_ + n];
                #pragma unroll
                for (int r = 0; r < 8; ++r) {
                    float v = cr.f[r] + bb;
                    v = v > 0.f ? v : 0.f;
                    p[mt][r] += v * w2;
                }
            }
        #pragma unroll
        for (int mt = 0; mt < 4; ++mt)
            #pragma unroll
            for (int r = 0; r < 8; ++r) {
                float v = p[mt][r];
                v += __shfl_xor(v, 1);
                v += __shfl_xor(v, 2);
                v += __shfl_xor(v, 4);
                v += __shfl_xor(v, 8);
                p[mt][r] = v;
            }
        if (lm == 0) {
            #pragma unroll
            for (int mt = 0; mt < 4; ++mt)
                #pragma unroll
                for (int r = 0; r < 8; ++r) {
                    const int ml = wm * 64 + mt * 16 + khalf * 8 + r;
                    sred[wn][ml] = p[mt][r];
                }
        }
        __syncthreads();
        {
            const int ml = tid;
            const float tot = sred[0][ml] + sred[1][ml] + b2[hop];
            const long orow = z * s_zoff + row0 + ml;
            sOut[orow] = accumulate ? (sOut[orow] + tot) : tot;
        }
        return;
    }

    // mode 0 / 1: emit bf16 hi/lo via LDS staging for coalesced stores.
    unsigned short* ldsS = (unsigned short*)sA_hi;   // reuse staging LDS (80KB)
    #pragma unroll
    for (int pass = 0; pass < 2; ++pass) {
        __syncthreads();
        #pragma unroll
        for (int mt = 0; mt < 4; ++mt)
            #pragma unroll
            for (int nt = 0; nt < 4; ++nt) {
                CReg cr; cr.v = acc[mt][nt];
                const int nl = wn * 64 + nt * 16 + lm;
                const int mb = wm * 64 + mt * 16 + khalf * 8;
                #pragma unroll
                for (int r = 0; r < 8; ++r) {
                    float v = cr.f[r];
                    unsigned short h = bf16_rne_bits(v);
                    unsigned short bits = pass ? bf16_rne_bits(v - bf16_bits_to_f(h)) : h;
                    ldsS[(mb + r) * ESTR + nl] = bits;
                }
            }
        __syncthreads();
        unsigned short* dst = (pass ? Clo : Chi);
        const unsigned short* src = ldsS + tid * ESTR;
        unsigned short* d = dst + (long)(row0 + tid) * ldcb + col0;
        #pragma unroll
        for (int c = 0; c < 16; ++c)
            *(uint4*)(d + c * 8) = *(const uint4*)(src + c * 8);
    }

    if (mode == 1) {
        // transposed bf16 hi/lo: stage [n][m] in LDS, write coalesced rows of C^T
        #pragma unroll
        for (int pass = 0; pass < 2; ++pass) {
            __syncthreads();
            #pragma unroll
            for (int mt = 0; mt < 4; ++mt)
                #pragma unroll
                for (int nt = 0; nt < 4; ++nt) {
                    CReg cr; cr.v = acc[mt][nt];
                    const int nl = wn * 64 + nt * 16 + lm;
                    const int mb = wm * 64 + mt * 16 + khalf * 8;
                    #pragma unroll
                    for (int r = 0; r < 8; ++r) {
                        float v = cr.f[r];
                        unsigned short h = bf16_rne_bits(v);
                        unsigned short bits = pass ? bf16_rne_bits(v - bf16_bits_to_f(h)) : h;
                        ldsS[nl * TSTR + mb + r] = bits;
                    }
                }
            __syncthreads();
            unsigned short* dst = (pass ? Ctlo : Cthi);
            const int n  = tid >> 1;
            const int mc = (tid & 1) * 128;
            const unsigned short* src = ldsS + n * TSTR + mc;
            unsigned short* d = dst + (long)(col0 + n) * ldct + row0 + mc;
            #pragma unroll
            for (int c = 0; c < 16; ++c)
                *(uint4*)(d + c * 8) = *(const uint4*)(src + c * 8);
        }
    }
}

// ------------------------------------------------------------------------------------
extern "C" void kernel_launch(void* const* d_in, const int* in_sizes, int n_in,
                              void* d_out, int out_size, void* d_ws, size_t ws_size,
                              hipStream_t stream) {
    const float* x  = (const float*)d_in[0];   // [B, W, M]
    const float* L  = (const float*)d_in[1];   // [M, M]
    const float* W1 = (const float*)d_in[2];   // [7, HID, W]
    const float* b1 = (const float*)d_in[3];   // [7, HID]
    const float* W2 = (const float*)d_in[4];   // [7, HID]
    const float* b2 = (const float*)d_in[5];   // [7]
    float* s = (float*)d_out;                  // [B, M]

    char* ws = (char*)d_ws;
    size_t off = 0;
    auto alloc = [&](size_t bytes) -> void* {
        void* p = ws + off;
        off = (off + bytes + 255) & ~(size_t)255;
        return p;
    };

    const size_t sz_bwm = (size_t)BW_ * M_ * 2;
    const size_t sz_mm  = (size_t)M_ * M_ * 2;
    const size_t sz_w1  = (size_t)(NL_ + 1) * HID_ * W_ * 2;

    unsigned short* xf_hi = (unsigned short*)alloc(sz_bwm);
    unsigned short* xf_lo = (unsigned short*)alloc(sz_bwm);
    unsigned short* xc_hi = (unsigned short*)alloc(sz_bwm);
    unsigned short* xc_lo = (unsigned short*)alloc(sz_bwm);
    unsigned short* L_hi  = (unsigned short*)alloc(sz_mm);
    unsigned short* L_lo  = (unsigned short*)alloc(sz_mm);
    unsigned short* LT_hi = (unsigned short*)alloc(sz_mm);
    unsigned short* LT_lo = (unsigned short*)alloc(sz_mm);
    unsigned short* Ph[2], *Pl[2], *PTh[2], *PTl[2];
    for (int k = 0; k < 2; ++k) {
        Ph[k]  = (unsigned short*)alloc(sz_mm);
        Pl[k]  = (unsigned short*)alloc(sz_mm);
        PTh[k] = (unsigned short*)alloc(sz_mm);
        PTl[k] = (unsigned short*)alloc(sz_mm);
    }
    unsigned short* T_hi  = (unsigned short*)alloc(sz_bwm);
    unsigned short* T_lo  = (unsigned short*)alloc(sz_bwm);
    unsigned short* W1_hi = (unsigned short*)alloc(sz_w1);
    unsigned short* W1_lo = (unsigned short*)alloc(sz_w1);

    // --- precompute bf16 hi/lo splits ---
    {
        int nx = BW_ * M_;
        k_split_x<<<(nx + 255) / 256, 256, 0, stream>>>(x, xf_hi, xf_lo, xc_hi, xc_lo);
        int nl = M_ * M_;
        k_split<<<(nl + 255) / 256, 256, 0, stream>>>(L, L_hi, L_lo, LT_hi, LT_lo, M_, M_);
        int nw = (NL_ + 1) * HID_ * W_;
        k_split<<<(nw + 255) / 256, 256, 0, stream>>>(W1, W1_hi, W1_lo, nullptr, nullptr,
                                                      (NL_ + 1) * HID_, W_);
    }

    // --- hop 0: fused MLP directly on xcat rows (t = xt), writes s ---
    dim3 gmlp(HID_ / BN, M_ / BM, B_);   // (1, 8, 32)
    k_gemm<true><<<gmlp, 256, 0, stream>>>(
        xc_hi, xc_lo, (long)BW_, (long)W_,
        W1_hi, W1_lo, (long)W_, W_, /*mode=*/2,
        nullptr, nullptr, 0L, nullptr, nullptr, 0L,
        b1, W2, b2, s, (long)M_, 0, 0);

    // --- hops 1..6 ---
    const unsigned short *cPh = L_hi,  *cPl = L_lo;     // P_1 = L
    const unsigned short *cPTh = LT_hi, *cPTl = LT_lo;
    for (int i = 1; i <= NL_; ++i) {
        // T = P_i @ Xcat, emitted directly as bf16 hi/lo (mode 0)
        dim3 g1(BW_ / BN, M_ / BM, 1);   // (42, 8, 1)
        k_gemm<false><<<g1, 256, 0, stream>>>(
            cPh, cPl, (long)M_, 0L,
            xf_hi, xf_lo, (long)M_, M_, /*mode=*/0,
            T_hi, T_lo, (long)BW_, nullptr, nullptr, 0L,
            b1, W2, b2, s, 0L, 0, 0);

        // fused MLP: s += W2 . relu(t_b @ W1[i]^T + b1) + b2
        k_gemm<true><<<gmlp, 256, 0, stream>>>(
            T_hi, T_lo, (long)BW_, (long)W_,
            W1_hi + (size_t)i * HID_ * W_, W1_lo + (size_t)i * HID_ * W_, (long)W_, W_, /*mode=*/2,
            nullptr, nullptr, 0L, nullptr, nullptr, 0L,
            b1, W2, b2, s, (long)M_, i, 1);

        // P_{i+1} = L @ P_i, emitted as bf16 hi/lo + transposed hi/lo (mode 1)
        if (i < NL_) {
            const int d = i & 1;
            dim3 g2(M_ / BN, M_ / BM, 1);   // (16, 8, 1)
            k_gemm<false><<<g2, 256, 0, stream>>>(
                L_hi, L_lo, (long)M_, 0L,
                cPTh, cPTl, (long)M_, M_, /*mode=*/1,
                Ph[d], Pl[d], (long)M_, PTh[d], PTl[d], (long)M_,
                b1, W2, b2, s, 0L, 0, 0);
            cPh = Ph[d]; cPl = Pl[d]; cPTh = PTh[d]; cPTl = PTl[d];
        }
    }
    (void)in_sizes; (void)n_in; (void)out_size; (void)ws_size;
}
